// TITANSMemoryModule_4767413698780
// MI455X (gfx1250) — matrix operand order repaired and checked by reading, not compiled, not run on hardware
//
#include <hip/hip_runtime.h>
#include <math.h>

typedef unsigned short u16;
typedef __bf16 v16bf __attribute__((ext_vector_type(16)));
typedef float  v8f   __attribute__((ext_vector_type(8)));

union FragBF { uint4 q[2]; v16bf v; };

#define Bc 4
#define Lc 2048
#define Dc 1024
#define Hc 256
#define Sc 32
#define FIFOc 512
#define NHc 4
#define HDc 256

__device__ __forceinline__ float bf2f(u16 h){ union{unsigned u; float f;} c; c.u=((unsigned)h)<<16; return c.f; }
__device__ __forceinline__ u16 f2bf(float f){ union{unsigned u; float f;} c; c.f=f; unsigned u=c.u; u += 0x7FFFu + ((u>>16)&1u); return (u16)(u>>16); }
__device__ __forceinline__ float geluf(float x){ return 0.5f*x*(1.f+erff(x*0.7071067811865476f)); }

/* ---- CDNA5 async global->LDS staging (ASYNCcnt-tracked) ---- */
__device__ __forceinline__ unsigned ldsOff(const void* p){
  return (unsigned)(unsigned long long)(__attribute__((address_space(3))) const void*)p;
}
__device__ __forceinline__ void async_ld_b128(unsigned ldsoff, const void* gptr){
  unsigned long long ga = (unsigned long long)(size_t)gptr;
  asm volatile("global_load_async_to_lds_b128 %0, %1, off"
               :: "v"(ldsoff), "v"(ga) : "memory");
}
__device__ __forceinline__ void wait_async0(){
  asm volatile("s_wait_asynccnt 0" ::: "memory");
}

__device__ __forceinline__ float block_sum(float v, float* sm){
  int tid=threadIdx.x, lane=tid&31, wid=tid>>5;
  #pragma unroll
  for(int o=16;o>0;o>>=1) v += __shfl_down(v,o,32);
  __syncthreads();
  if(lane==0) sm[wid]=v;
  __syncthreads();
  float r = (tid < ((int)blockDim.x>>5)) ? sm[tid] : 0.f;
  if(wid==0){
    #pragma unroll
    for(int o=16;o>0;o>>=1) r += __shfl_down(r,o,32);
    if(lane==0) sm[32]=r;
  }
  __syncthreads();
  return sm[32];
}
__device__ __forceinline__ float block_maxf(float v, float* sm){
  int tid=threadIdx.x, lane=tid&31, wid=tid>>5;
  #pragma unroll
  for(int o=16;o>0;o>>=1) v = fmaxf(v,__shfl_down(v,o,32));
  __syncthreads();
  if(lane==0) sm[wid]=v;
  __syncthreads();
  float r = (tid < ((int)blockDim.x>>5)) ? sm[tid] : -3.4e38f;
  if(wid==0){
    #pragma unroll
    for(int o=16;o>0;o>>=1) r = fmaxf(r,__shfl_down(r,o,32));
    if(lane==0) sm[32]=r;
  }
  __syncthreads();
  return sm[32];
}

/* ------------------------------------------------------------------ */
/* Generic strided-batched WMMA bf16 GEMM: C = act(scale*A@W^T + bias) */
/* A: (M,K) lda; W: (N,K) ldw; C: (M,N) ldc. M % 128 == 0, K % 32 == 0 */
/* ------------------------------------------------------------------ */
#define BM 128
#define BN 128
#define BK 32
#define LDT 40

__global__ __launch_bounds__(256)
void wmma_gemm(const u16* __restrict__ A, const u16* __restrict__ W,
               const float* __restrict__ bias, float* __restrict__ Cf,
               u16* __restrict__ Cb, const float* __restrict__ Res,
               int M, int N, int K, int lda, int ldw, int ldc,
               long long sAb, long long sAh, long long sWb, long long sWh,
               long long sCbat, long long sCh, int nH,
               float scale, int do_gelu)
{
  __shared__ u16 As[2][BM*LDT];
  __shared__ u16 Bs[2][BM*LDT];

  int z = blockIdx.z;
  int bb = z / nH, hh = z % nH;
  size_t offA = (size_t)bb*sAb + (size_t)hh*sAh;
  size_t offW = (size_t)bb*sWb + (size_t)hh*sWh;
  size_t offC = (size_t)bb*sCbat + (size_t)hh*sCh;

  int m0 = blockIdx.y*BM, n0 = blockIdx.x*BN;
  int tid=threadIdx.x, lane=tid&31, wid=tid>>5;
  int wm = (wid>>2)*64;
  int wn = (wid&3)*32;
  int l16 = lane&15; int hi = lane>>4;

  v8f acc[4][2];
  #pragma unroll
  for(int i=0;i<4;i++)
    #pragma unroll
    for(int j=0;j<2;j++)
      #pragma unroll
      for(int e=0;e<8;e++) acc[i][j][e]=0.f;

  bool bfull = (n0 + BM) <= N;   // uniform: whole B tile in-bounds

  auto stage = [&](int kt, int buf){
    int k0 = kt*BK;
    #pragma unroll
    for(int r=0;r<2;r++){
      int cid = r*256 + tid; int row = cid>>2; int c8 = (cid&3)*8;
      // A tile: always in-bounds (M % 128 == 0) -> pure async direct-to-LDS
      async_ld_b128(ldsOff(&As[buf][row*LDT + c8]),
                    A + offA + (size_t)(m0+row)*lda + k0 + c8);
      // B tile: async when in-bounds, zero-fill via DS store otherwise
      if(bfull || (n0+row) < N){
        async_ld_b128(ldsOff(&Bs[buf][row*LDT + c8]),
                      W + offW + (size_t)(n0+row)*ldw + k0 + c8);
      } else {
        uint4 zz; zz.x=zz.y=zz.z=zz.w=0u;
        *(uint4*)(void*)&Bs[buf][row*LDT + c8] = zz;
      }
    }
  };

  int nkt = K/BK;
  stage(0,0);
  wait_async0();
  __syncthreads();

  for(int kt=0; kt<nkt; ++kt){
    int curb = kt&1;
    if(kt+1<nkt) stage(kt+1, curb^1);
    if(kt+2<nkt){
      // CDNA5 global_prefetch_b8 of the K+2 tile
      __builtin_prefetch(A + offA + (size_t)(m0 + (tid>>2))*lda + (kt+2)*BK + (tid&3)*8, 0, 1);
      __builtin_prefetch(W + offW + (size_t)(n0 + (tid>>2))*ldw + (kt+2)*BK + (tid&3)*8, 0, 1);
    }
    FragBF af[4], bfr[2];
    int aof = hi ? 8 : 0;
    #pragma unroll
    for(int mt=0;mt<4;mt++){
      int row = wm + mt*16 + l16;
      const u16* p = &As[curb][row*LDT + aof];
      af[mt].q[0] = *(const uint4*)(const void*)p;
      af[mt].q[1] = *(const uint4*)(const void*)(p+16);
    }
    int bof = hi ? 8 : 0;
    #pragma unroll
    for(int nt=0;nt<2;nt++){
      int row = wn + nt*16 + l16;
      const u16* p = &Bs[curb][row*LDT + bof];
      bfr[nt].q[0] = *(const uint4*)(const void*)p;
      bfr[nt].q[1] = *(const uint4*)(const void*)(p+16);
    }
    #pragma unroll
    for(int mt=0;mt<4;mt++)
      #pragma unroll
      for(int nt=0;nt<2;nt++)
        acc[mt][nt] = __builtin_amdgcn_wmma_f32_16x16x32_bf16(
            false, af[mt].v, false, bfr[nt].v, (short)0, acc[mt][nt], false, false);
    wait_async0();
    __syncthreads();
  }

  #pragma unroll
  for(int nt=0;nt<2;nt++){
    int n = n0 + wn + nt*16 + l16;
    if(n >= N) continue;
    float bv = bias ? bias[n] : 0.f;
    #pragma unroll
    for(int mt=0;mt<4;mt++){
      int mb = m0 + wm + mt*16 + hi*8;
      #pragma unroll
      for(int r=0;r<8;r++){
        float v = acc[mt][nt][r]*scale + bv;
        if(do_gelu) v = 0.5f*v*(1.f+erff(v*0.7071067811865476f));
        size_t ci = offC + (size_t)(mb+r)*ldc + n;
        if(Res) v += Res[ci];
        if(Cf) Cf[ci] = v;
        if(Cb) Cb[ci] = f2bf(v);
      }
    }
  }
}

/* ---------------------------- small kernels ---------------------------- */

__global__ void cast_f2bf(const float* __restrict__ in, u16* __restrict__ out, int n){
  int i = blockIdx.x*256 + threadIdx.x;
  if(i<n) out[i] = f2bf(in[i]);
}

__global__ void prep_slots(const float* __restrict__ slots, u16* __restrict__ snorm_bf,
                           u16* __restrict__ slotsT_bf){
  __shared__ float sm[33];
  int s = blockIdx.x, tid = threadIdx.x;
  float v[4]; float ss=0.f;
  #pragma unroll
  for(int j=0;j<4;j++){ v[j]=slots[s*Dc + tid + j*256]; ss += v[j]*v[j]; }
  ss = block_sum(ss, sm);
  float inv = 1.f / fmaxf(sqrtf(ss), 1e-12f);
  #pragma unroll
  for(int j=0;j<4;j++){
    int d = tid + j*256;
    snorm_bf[s*Dc + d] = f2bf(v[j]*inv);
    slotsT_bf[(size_t)d*Sc + s] = f2bf(v[j]);
  }
}

__global__ void rmsnorm_kernel(const float* __restrict__ x, const float* __restrict__ w,
                               float* __restrict__ xn_f, u16* __restrict__ xn_bf,
                               u16* __restrict__ qn_bf, float* __restrict__ fifo_out){
  __shared__ float sm[33];
  int row = blockIdx.x; int b = row / Lc; int l = row % Lc;
  int tid = threadIdx.x;
  size_t base = (size_t)row*Dc;
  float xv[4]; float ss=0.f;
  #pragma unroll
  for(int j=0;j<4;j++){ xv[j]=x[base+tid+j*256]; ss += xv[j]*xv[j]; }
  ss = block_sum(ss, sm);
  float rinv = rsqrtf(ss/(float)Dc + 1.1920929e-07f);
  float xnv[4]; float s2=0.f;
  #pragma unroll
  for(int j=0;j<4;j++){ xnv[j]=xv[j]*rinv*w[tid+j*256]; s2 += xnv[j]*xnv[j]; }
  s2 = block_sum(s2, sm);
  float q = 1.f / fmaxf(sqrtf(s2), 1e-12f);
  #pragma unroll
  for(int j=0;j<4;j++){
    int d = tid + j*256;
    xn_f[base+d]  = xnv[j];
    xn_bf[base+d] = f2bf(xnv[j]);
    qn_bf[base+d] = f2bf(xnv[j]*q);
  }
  if(l >= Lc - FIFOc){
    size_t fo = ((size_t)b*FIFOc + (l - (Lc - FIFOc)))*Dc;
    #pragma unroll
    for(int j=0;j<4;j++) fifo_out[fo + tid + j*256] = xnv[j];
  }
}

__global__ void surprise_kernel(const float* __restrict__ expected, const float* __restrict__ lt,
                                float* __restrict__ mask){
  __shared__ float sm[33];
  int row = blockIdx.x, tid = threadIdx.x;
  size_t base = (size_t)row*Dc;
  float acc=0.f;
  #pragma unroll
  for(int j=0;j<4;j++){ float d = expected[base+tid+j*256]-lt[base+tid+j*256]; acc += d*d; }
  acc = block_sum(acc, sm);
  if(tid==0) mask[row] = (acc/(float)Dc > 0.1f) ? 1.f : 0.f;
}

__global__ void cnt_kernel(const float* __restrict__ mask, float* __restrict__ cnt){
  __shared__ float sm[33];
  int b = blockIdx.x, tid = threadIdx.x;
  float s=0.f;
  for(int l=tid; l<Lc; l+=256) s += mask[b*Lc+l];
  s = block_sum(s, sm);
  if(tid==0) cnt[b]=s;
}

__global__ void wb_kernel(const float* __restrict__ xn_f, const float* __restrict__ mask,
                          const float* __restrict__ cnt, float* __restrict__ wb){
  int b = blockIdx.x; int d = blockIdx.y*256 + threadIdx.x;
  float s=0.f;
  for(int l=0;l<Lc;l++) s += xn_f[((size_t)b*Lc+l)*Dc + d]*mask[b*Lc+l];
  wb[b*Dc+d] = s / fmaxf(cnt[b], 1.f);
}

__global__ void grad_ha(const float* __restrict__ wb, const float* __restrict__ W1,
                        float* __restrict__ h_f, float* __restrict__ a_f){
  int b = blockIdx.x, j = threadIdx.x;
  const float* w = W1 + (size_t)j*Dc; const float* t = wb + b*Dc;
  float s=0.f;
  for(int d=0; d<Dc; d++) s += t[d]*w[d];
  h_f[b*Hc+j]=s; a_f[b*Hc+j]=geluf(s);
}

__global__ void grad_rdr(const float* __restrict__ a_f, const float* __restrict__ W2,
                         const float* __restrict__ wb, const float* __restrict__ cnt,
                         float* __restrict__ dr_f){
  int b = blockIdx.x, tid = threadIdx.x;
  float nval=0.f;
  for(int i=0;i<Bc;i++) nval += (cnt[i]>0.f)?1.f:0.f;
  float denom = fmaxf(nval,1.f)*(float)Dc;
  bool valid = cnt[b]>0.f;
  for(int jj=0;jj<4;jj++){
    int n = tid + jj*256;
    const float* w = W2 + (size_t)n*Hc; const float* av = a_f + b*Hc;
    float s=0.f;
    for(int k=0;k<Hc;k++) s += av[k]*w[k];
    dr_f[b*Dc+n] = valid ? 2.f*(s - wb[b*Dc+n])/denom : 0.f;
  }
}

__global__ void w2_update(const float* __restrict__ dr_f, const float* __restrict__ a_f,
                          const float* __restrict__ W2, float* __restrict__ outW2){
  int idx = blockIdx.x*256 + threadIdx.x;   // (1024,256) flat
  int n = idx>>8, j = idx&255;
  float g=0.f;
  for(int b=0;b<Bc;b++) g += dr_f[b*Dc+n]*a_f[b*Hc+j];
  outW2[idx] = W2[idx] - 0.01f*g;
}

__global__ void grad_dh(const float* __restrict__ dr_f, const float* __restrict__ W2,
                        const float* __restrict__ h_f, float* __restrict__ dh_f){
  int b = blockIdx.x, j = threadIdx.x;
  float s=0.f;
  for(int n=0;n<Dc;n++) s += dr_f[b*Dc+n]*W2[(size_t)n*Hc+j];
  float xx = h_f[b*Hc+j];
  float dg = 0.5f*(1.f+erff(xx*0.7071067811865476f)) + xx*expf(-0.5f*xx*xx)*0.3989422804014327f;
  dh_f[b*Hc+j] = s*dg;
}

__global__ void w1_update(const float* __restrict__ dh_f, const float* __restrict__ wb,
                          const float* __restrict__ W1, float* __restrict__ outW1){
  int idx = blockIdx.x*256 + threadIdx.x;   // (256,1024) flat
  int j = idx>>10, d = idx&1023;
  float g=0.f;
  for(int b=0;b<Bc;b++) g += dh_f[b*Hc+j]*wb[b*Dc+d];
  outW1[idx] = W1[idx] - 0.01f*g;
}

__global__ __launch_bounds__(1024)
void summary_scan(const float* __restrict__ slots_in, const float* __restrict__ usage_in,
                  const float* __restrict__ wb, const float* __restrict__ cnt,
                  float* __restrict__ slots_out, float* __restrict__ usage_out){
  __shared__ float sims[Sc]; __shared__ float sn2[Sc]; __shared__ float tok2s;
  __shared__ int sidx;
  int tid=threadIdx.x, wid=tid>>5, lane=tid&31;
  for(int i=tid;i<Sc*Dc;i+=1024) slots_out[i]=slots_in[i];
  if(tid<Sc) usage_out[tid]=usage_in[tid];
  __syncthreads();
  for(int t=0;t<Bc;t++){
    bool ok = cnt[t] > 0.f;
    const float* tok = wb + t*Dc;
    float nn=0.f, dd=0.f;
    for(int d=lane; d<Dc; d+=32){ float sv=slots_out[wid*Dc+d]; float tv=tok[d]; nn+=sv*sv; dd+=sv*tv; }
    #pragma unroll
    for(int o=16;o>0;o>>=1){ nn+=__shfl_down(nn,o,32); dd+=__shfl_down(dd,o,32); }
    if(lane==0){ sn2[wid]=nn; sims[wid]=dd; }
    if(wid==0){
      float tq=0.f;
      for(int d=lane; d<Dc; d+=32){ float tv=tok[d]; tq+=tv*tv; }
      #pragma unroll
      for(int o=16;o>0;o>>=1) tq+=__shfl_down(tq,o,32);
      if(lane==0) tok2s=tq;
    }
    __syncthreads();
    if(tid==0){
      float tnorm = fmaxf(sqrtf(tok2s), 1e-8f);
      float usum=0.f; for(int s=0;s<Sc;s++) usum += usage_out[s];
      int idx=0;
      if(usum>0.f){
        int eidx=-1;
        for(int s=0;s<Sc;s++) if(usage_out[s]==0.f){ eidx=s; break; }
        if(eidx>=0) idx=eidx;
        else{
          float best=-3.4e38f;
          for(int s=0;s<Sc;s++){
            float sv = sims[s]/(fmaxf(sqrtf(sn2[s]),1e-8f)*tnorm);
            if(sv>best){ best=sv; idx=s; }
          }
        }
      }
      sidx=idx;
    }
    __syncthreads();
    int idx=sidx;
    if(ok){
      slots_out[idx*Dc+tid] = 0.95f*slots_out[idx*Dc+tid] + 0.05f*tok[tid];
      if(tid==0) usage_out[idx] += 1.f;
    }
    __syncthreads();
  }
}

__global__ void softmax32(const float* __restrict__ sims, u16* __restrict__ sw){
  int tid=threadIdx.x, lane=tid&31;
  int r = blockIdx.x*8 + (tid>>5);
  float v = sims[(size_t)r*Sc + lane];
  float m=v;
  #pragma unroll
  for(int o=16;o>0;o>>=1) m = fmaxf(m,__shfl_xor(m,o,32));
  float e = expf(v-m); float s=e;
  #pragma unroll
  for(int o=16;o>0;o>>=1) s += __shfl_xor(s,o,32);
  sw[(size_t)r*Sc + lane] = f2bf(e/s);
}

__global__ void softmax512_bf16(u16* __restrict__ p){
  __shared__ float sm[33];
  size_t row = blockIdx.x;
  u16* pr = p + row*512;
  int tid = threadIdx.x;
  float v0=bf2f(pr[tid]), v1=bf2f(pr[tid+256]);
  float mx = block_maxf(fmaxf(v0,v1), sm);
  float e0=expf(v0-mx), e1=expf(v1-mx);
  float s = block_sum(e0+e1, sm);
  float inv = 1.f/s;
  pr[tid]     = f2bf(e0*inv);
  pr[tid+256] = f2bf(e1*inv);
}

__global__ void transpose_v(const u16* __restrict__ v, u16* __restrict__ vT){
  int i = blockIdx.x*256 + threadIdx.x;          // B*NH*HD*512
  int kk = i & 511; int n = (i>>9)&255; int h=(i>>17)&3; int b=i>>19;
  vT[i] = v[ ((size_t)(b*FIFOc+kk))*Dc + h*HDc + n ];
}

__global__ void fusion_kernel(const float* __restrict__ xn, const float* __restrict__ st,
                              const float* __restrict__ lt, const float* __restrict__ sum_r,
                              const float* __restrict__ fw, const float* __restrict__ fb,
                              const float* __restrict__ usage, u16* __restrict__ fused_bf){
  __shared__ float red[3][8]; __shared__ float g[4];
  int row = blockIdx.x, tid = threadIdx.x, wid=tid>>5, lane=tid&31;
  if(tid==0){ float s=0.f; for(int i=0;i<Sc;i++) s+=usage[i]; g[3] = (s>0.f)?1.f:0.f; }
  __syncthreads();
  float flag = g[3];
  size_t base=(size_t)row*Dc;
  float p0=0.f,p1=0.f,p2=0.f;
  #pragma unroll
  for(int j=0;j<4;j++){
    int d = tid + j*256;
    float a=xn[base+d], bv=st[base+d], c=lt[base+d], e=sum_r[base+d]*flag;
    p0 += a*fw[d]      + bv*fw[1024+d]      + c*fw[2048+d]      + e*fw[3072+d];
    p1 += a*fw[4096+d] + bv*fw[4096+1024+d] + c*fw[4096+2048+d] + e*fw[4096+3072+d];
    p2 += a*fw[8192+d] + bv*fw[8192+1024+d] + c*fw[8192+2048+d] + e*fw[8192+3072+d];
  }
  #pragma unroll
  for(int o=16;o>0;o>>=1){ p0+=__shfl_down(p0,o,32); p1+=__shfl_down(p1,o,32); p2+=__shfl_down(p2,o,32); }
  if(lane==0){ red[0][wid]=p0; red[1][wid]=p1; red[2][wid]=p2; }
  __syncthreads();
  if(tid==0){
    float l0=fb[0],l1=fb[1],l2=fb[2];
    for(int w=0;w<8;w++){ l0+=red[0][w]; l1+=red[1][w]; l2+=red[2][w]; }
    float m=fmaxf(l0,fmaxf(l1,l2));
    float e0=expf(l0-m),e1=expf(l1-m),e2=expf(l2-m); float s=e0+e1+e2;
    g[0]=e0/s; g[1]=e1/s; g[2]=e2/s;
  }
  __syncthreads();
  float g0=g[0],g1=g[1],g2=g[2];
  #pragma unroll
  for(int j=0;j<4;j++){
    int d = tid + j*256;
    float v = g0*st[base+d] + g1*lt[base+d] + g2*(sum_r[base+d]*flag);
    fused_bf[base+d]=f2bf(v);
  }
}

/* ---------------------------- host launcher ---------------------------- */

static void gemm(hipStream_t s, const u16* A, const u16* W, const float* bias,
                 float* Cf, u16* Cb, const float* Res,
                 int M,int N,int K,int lda,int ldw,int ldc,
                 long long sAb,long long sAh,long long sWb,long long sWh,
                 long long sCb,long long sCh,int nH,int nbatch,float scale,int gl)
{
  dim3 g((N+BN-1)/BN, M/BM, nbatch);
  wmma_gemm<<<g, 256, 0, s>>>(A,W,bias,Cf,Cb,Res,M,N,K,lda,ldw,ldc,
                              sAb,sAh,sWb,sWh,sCb,sCh,nH,scale,gl);
}

extern "C" void kernel_launch(void* const* d_in, const int* in_sizes, int n_in,
                              void* d_out, int out_size, void* d_ws, size_t ws_size,
                              hipStream_t stream){
  (void)in_sizes; (void)n_in; (void)out_size; (void)ws_size;
  const float* x    = (const float*)d_in[0];
  const float* fifo = (const float*)d_in[1];
  const float* nwt  = (const float*)d_in[2];
  const float* W1   = (const float*)d_in[3];
  const float* W2   = (const float*)d_in[4];
  const float* sgw1 = (const float*)d_in[5];
  const float* sgw2 = (const float*)d_in[6];
  const float* inpw = (const float*)d_in[7];
  const float* inpb = (const float*)d_in[8];
  const float* mow  = (const float*)d_in[9];
  const float* mob  = (const float*)d_in[10];
  const float* fw   = (const float*)d_in[11];
  const float* fb   = (const float*)d_in[12];
  const float* opw  = (const float*)d_in[13];
  const float* slots= (const float*)d_in[14];
  const float* usage= (const float*)d_in[15];

  float* out       = (float*)d_out;
  float* out_fifo  = out + 8388608;
  float* out_W1    = out + 10485760;
  float* out_W2    = out + 10747904;
  float* out_slots = out + 11010048;
  float* out_usage = out + 11042816;

  char* wsb=(char*)d_ws; size_t cur=0;
  auto alloc=[&](size_t bytes)->void*{ void* p=wsb+cur; cur=(cur+bytes+255)&~(size_t)255; return p; };

  const size_t TOK = (size_t)Bc*Lc*Dc;           // 8388608
  float* xn_f   = (float*)alloc(TOK*4);
  float* lt_f   = (float*)alloc(TOK*4);
  float* exp_f  = (float*)alloc(TOK*4);          // reused as st_recall
  float* sum_f  = (float*)alloc(TOK*4);
  float* sims_f = (float*)alloc((size_t)Bc*Lc*Sc*4);
  float* mask   = (float*)alloc((size_t)Bc*Lc*4);
  float* cnt    = (float*)alloc(Bc*4);
  float* wb     = (float*)alloc(Bc*Dc*4);
  float* h_f    = (float*)alloc(Bc*Hc*4);
  float* a_f    = (float*)alloc(Bc*Hc*4);
  float* dr_f   = (float*)alloc(Bc*Dc*4);
  float* dh_f   = (float*)alloc(Bc*Hc*4);

  u16* xn_bf   = (u16*)alloc(TOK*2);             // reused as attn_bf
  u16* qn_bf   = (u16*)alloc(TOK*2);             // reused as fused_bf
  u16* h1_bf   = (u16*)alloc((size_t)Bc*Lc*Hc*2);
  u16* e1_bf   = (u16*)alloc(TOK*2);             // reused as q_bf
  u16* k_bf    = (u16*)alloc((size_t)Bc*FIFOc*Dc*2);
  u16* v_bf    = (u16*)alloc((size_t)Bc*FIFOc*Dc*2);
  u16* vT_bf   = (u16*)alloc((size_t)Bc*NHc*HDc*512*2);
  u16* probs   = (u16*)alloc((size_t)Bc*NHc*Lc*512*2);
  u16* fifo_bf = (u16*)alloc((size_t)Bc*FIFOc*Dc*2);
  u16* sw_bf   = (u16*)alloc((size_t)Bc*Lc*Sc*2);
  u16* w1_bf   = (u16*)alloc((size_t)Hc*Dc*2);
  u16* w2_bf   = (u16*)alloc((size_t)Dc*Hc*2);
  u16* sgw1_bf = (u16*)alloc((size_t)Dc*Dc*2);
  u16* sgw2_bf = (u16*)alloc((size_t)Dc*Dc*2);
  u16* wqkv_bf = (u16*)alloc((size_t)3*Dc*Dc*2);
  u16* mow_bf  = (u16*)alloc((size_t)Dc*Dc*2);
  u16* opw_bf  = (u16*)alloc((size_t)Dc*Dc*2);
  u16* slT_bf  = (u16*)alloc((size_t)Dc*Sc*2);
  u16* sn_bf   = (u16*)alloc((size_t)Sc*Dc*2);

  u16* attn_bf  = xn_bf;
  u16* fused_bf = qn_bf;
  u16* q_bf     = e1_bf;
  float* st_f   = exp_f;

  // 1. weight / fifo casts
  auto cast = [&](const float* src, u16* dst, int n){
    cast_f2bf<<<(n+255)/256, 256, 0, stream>>>(src, dst, n);
  };
  cast(W1,   w1_bf,   Hc*Dc);
  cast(W2,   w2_bf,   Dc*Hc);
  cast(sgw1, sgw1_bf, Dc*Dc);
  cast(sgw2, sgw2_bf, Dc*Dc);
  cast(inpw, wqkv_bf, 3*Dc*Dc);
  cast(mow,  mow_bf,  Dc*Dc);
  cast(opw,  opw_bf,  Dc*Dc);
  cast(fifo, fifo_bf, Bc*FIFOc*Dc);
  prep_slots<<<Sc, 256, 0, stream>>>(slots, sn_bf, slT_bf);

  // 2. RMSNorm (+qn, +new_fifo)
  rmsnorm_kernel<<<Bc*Lc, 256, 0, stream>>>(x, nwt, xn_f, xn_bf, qn_bf, out_fifo);

  // 3. memory MLP + surprise MLP + cosine sims
  gemm(stream, xn_bf, w1_bf,   nullptr, nullptr, h1_bf, nullptr, Bc*Lc, Hc, Dc, Dc, Dc, Hc,
       0,0,0,0,0,0, 1,1, 1.f, 1);
  gemm(stream, h1_bf, w2_bf,   nullptr, lt_f,   nullptr, nullptr, Bc*Lc, Dc, Hc, Hc, Hc, Dc,
       0,0,0,0,0,0, 1,1, 1.f, 0);
  gemm(stream, xn_bf, sgw1_bf, nullptr, nullptr, e1_bf, nullptr, Bc*Lc, Dc, Dc, Dc, Dc, Dc,
       0,0,0,0,0,0, 1,1, 1.f, 1);
  gemm(stream, e1_bf, sgw2_bf, nullptr, exp_f,  nullptr, nullptr, Bc*Lc, Dc, Dc, Dc, Dc, Dc,
       0,0,0,0,0,0, 1,1, 1.f, 0);
  gemm(stream, qn_bf, sn_bf,   nullptr, sims_f, nullptr, nullptr, Bc*Lc, Sc, Dc, Dc, Dc, Sc,
       0,0,0,0,0,0, 1,1, 1.f/0.35f, 0);

  // 4. surprise -> mask -> write batch
  surprise_kernel<<<Bc*Lc, 256, 0, stream>>>(exp_f, lt_f, mask);
  cnt_kernel<<<Bc, 256, 0, stream>>>(mask, cnt);
  wb_kernel<<<dim3(Bc, Dc/256), 256, 0, stream>>>(xn_f, mask, cnt, wb);

  // 5. inner gradient step (writes new_W1/new_W2 to d_out)
  grad_ha  <<<Bc, 256, 0, stream>>>(wb, W1, h_f, a_f);
  grad_rdr <<<Bc, 256, 0, stream>>>(a_f, W2, wb, cnt, dr_f);
  w2_update<<<(Dc*Hc)/256, 256, 0, stream>>>(dr_f, a_f, W2, out_W2);
  grad_dh  <<<Bc, 256, 0, stream>>>(dr_f, W2, h_f, dh_f);
  w1_update<<<(Hc*Dc)/256, 256, 0, stream>>>(dh_f, wb, W1, out_W1);

  // 6. sequential slot EMA scan (writes new_slots/new_usage)
  summary_scan<<<1, 1024, 0, stream>>>(slots, usage, wb, cnt, out_slots, out_usage);

  // 7. summary retrieval
  softmax32<<<(Bc*Lc)/8, 256, 0, stream>>>(sims_f, sw_bf);
  gemm(stream, sw_bf, slT_bf, nullptr, sum_f, nullptr, nullptr, Bc*Lc, Dc, Sc, Sc, Sc, Dc,
       0,0,0,0,0,0, 1,1, 1.f, 0);

  // 8. MHA: q/k/v projections
  gemm(stream, xn_bf,   wqkv_bf,              inpb,        nullptr, q_bf, nullptr,
       Bc*Lc,    Dc, Dc, Dc, Dc, Dc, 0,0,0,0,0,0, 1,1, 1.f, 0);
  gemm(stream, fifo_bf, wqkv_bf + 1048576,    inpb + 1024, nullptr, k_bf, nullptr,
       Bc*FIFOc, Dc, Dc, Dc, Dc, Dc, 0,0,0,0,0,0, 1,1, 1.f, 0);
  gemm(stream, fifo_bf, wqkv_bf + 2097152,    inpb + 2048, nullptr, v_bf, nullptr,
       Bc*FIFOc, Dc, Dc, Dc, Dc, Dc, 0,0,0,0,0,0, 1,1, 1.f, 0);

  // 9. scores = q@k^T / sqrt(HD), batched over (b,h)
  gemm(stream, q_bf, k_bf, nullptr, nullptr, probs, nullptr,
       Lc, 512, HDc, Dc, Dc, 512,
       (long long)Lc*Dc, HDc, (long long)FIFOc*Dc, HDc,
       (long long)NHc*Lc*512, (long long)Lc*512, NHc, Bc*NHc, 0.0625f, 0);
  softmax512_bf16<<<Bc*NHc*Lc, 256, 0, stream>>>(probs);

  // 10. attn = probs @ v  (via v^T), batched
  transpose_v<<<(Bc*NHc*HDc*512)/256, 256, 0, stream>>>(v_bf, vT_bf);
  gemm(stream, probs, vT_bf, nullptr, nullptr, attn_bf, nullptr,
       Lc, HDc, 512, 512, 512, Dc,
       (long long)NHc*Lc*512, (long long)Lc*512,
       (long long)NHc*HDc*512, (long long)HDc*512,
       (long long)Lc*Dc, HDc, NHc, Bc*NHc, 1.f, 0);

  // 11. MHA out-proj
  gemm(stream, attn_bf, mow_bf, mob, st_f, nullptr, nullptr, Bc*Lc, Dc, Dc, Dc, Dc, Dc,
       0,0,0,0,0,0, 1,1, 1.f, 0);

  // 12. fusion gates -> fused (bf16)
  fusion_kernel<<<Bc*Lc, 256, 0, stream>>>(xn_f, st_f, lt_f, sum_f, fw, fb, usage, fused_bf);

  // 13. output = fused @ out_proj^T + x
  gemm(stream, fused_bf, opw_bf, nullptr, out, nullptr, x, Bc*Lc, Dc, Dc, Dc, Dc, Dc,
       0,0,0,0,0,0, 1,1, 1.f, 0);
}
